// SamVisionAttentionLayerBetterTransformer_66056597012548
// MI455X (gfx1250) — matrix operand order repaired and checked by reading, not compiled, not run on hardware
//
#include <hip/hip_runtime.h>
#include <hip/hip_bf16.h>

// ---------------------------------------------------------------------------
// Problem constants
// ---------------------------------------------------------------------------
#define BD    400
#define HWD   196          // 14*14
#define CC    768
#define NHD   12
#define HDIM  64
#define MTOT  (BD * HWD)   // 78400
#define BH    (BD * NHD)   // 4800

typedef __attribute__((ext_vector_type(16))) __bf16 v16bf;
typedef __attribute__((ext_vector_type(8)))  __bf16 v8bf;
typedef __attribute__((ext_vector_type(4)))  __bf16 v4bf;
typedef __attribute__((ext_vector_type(8)))  float  v8f;
typedef __attribute__((ext_vector_type(4)))  int    v4i;

static __device__ __forceinline__ v16bf ld_frag(const __bf16* p0, const __bf16* p1) {
    v8bf lo = *(const v8bf*)p0;
    v8bf hi = *(const v8bf*)p1;
    return __builtin_shufflevector(lo, hi, 0,1,2,3,4,5,6,7,8,9,10,11,12,13,14,15);
}

static __device__ __forceinline__ v8f wmma_bf16(v16bf a, v16bf b, v8f c) {
    return __builtin_amdgcn_wmma_f32_16x16x32_bf16(false, a, false, b, (short)0, c,
                                                   false, false);
}

// ---------------------------------------------------------------------------
// CDNA5 async global->LDS staging (ASYNCcnt path), with safe fallback.
// The builtin expects (v4i addrspace(1)*, v4i addrspace(3)*, imm, imm).
// AS1 pointer: inttoptr of the 64-bit generic address (identical bits).
// AS3 pointer: inttoptr of addr[31:0] (ISA aperture rule: LDS offset).
// ---------------------------------------------------------------------------
#if defined(__AMDGCN__) && __has_builtin(__builtin_amdgcn_global_load_async_to_lds_b128)
#define HAS_ASYNC_LDS 1
#else
#define HAS_ASYNC_LDS 0
#endif

typedef __attribute__((address_space(1))) v4i gbl_v4i;
typedef __attribute__((address_space(3))) v4i lds_v4i;

static __device__ __forceinline__ void async_cp16(const void* g, void* l) {
#if HAS_ASYNC_LDS
    __builtin_amdgcn_global_load_async_to_lds_b128(
        (gbl_v4i*)(unsigned long long)(uintptr_t)g,
        (lds_v4i*)(unsigned)(uintptr_t)l,
        0, 0);
#else
    *(uint4*)l = *(const uint4*)g;
#endif
}

static __device__ __forceinline__ void wait_async() {
#if HAS_ASYNC_LDS
#if __has_builtin(__builtin_amdgcn_s_wait_asynccnt)
    __builtin_amdgcn_s_wait_asynccnt(0);
#else
    asm volatile("s_wait_asynccnt 0x0" ::: "memory");
#endif
#endif
}

// ---------------------------------------------------------------------------
// f32 -> bf16 conversion, 4 elements / thread
// ---------------------------------------------------------------------------
__global__ __launch_bounds__(256)
void cvt_f32_bf16(const float* __restrict__ s, __bf16* __restrict__ d, long long n4) {
    long long i = (long long)blockIdx.x * 256 + threadIdx.x;
    if (i >= n4) return;
    float4 v = ((const float4*)s)[i];
    v4bf o;
    o.x = (__bf16)v.x; o.y = (__bf16)v.y; o.z = (__bf16)v.z; o.w = (__bf16)v.w;
    ((v4bf*)d)[i] = o;
}

// ---------------------------------------------------------------------------
// QKV GEMM: X(78400x768,bf16) @ W(2304x768,bf16 N-major)^T + bias
// Block = 256 thr (8 waves), tile 64(M) x 256(N), K step 32, double-buffered
// async LDS staging. Wave tile 32x64 -> 8 accumulators, B-fragment reuse x2.
// Epilogue scatters to q/k/v buffers laid out (b, nh, hw, hd) in bf16.
// ---------------------------------------------------------------------------
__global__ __launch_bounds__(256)
void qkv_gemm(const __bf16* __restrict__ X, const __bf16* __restrict__ W,
              const float* __restrict__ bias,
              __bf16* __restrict__ qb, __bf16* __restrict__ kb, __bf16* __restrict__ vb) {
    __shared__ __bf16 As[2][64][32];
    __shared__ __bf16 Bs[2][256][32];
    const int tid  = threadIdx.x;
    const int wave = tid >> 5, lane = tid & 31;
    const int half = lane >> 4, l16 = lane & 15;
    const int m0 = (blockIdx.x % 1225) * 64;
    const int n0 = (blockIdx.x / 1225) * 256;
    const int wm = wave >> 2, wn = wave & 3;

    v8f acc[2][4] = {};

    auto stage = [&](int sb, int k0) {
        {   // A tile 64x32: one 16B chunk per thread
            const int r = tid >> 2, c = (tid & 3) * 8;
            async_cp16(&X[(size_t)(m0 + r) * 768 + k0 + c], &As[sb][r][c]);
        }
        {   // B tile 256x32: one full row (4 x 16B) per thread
            const __bf16* src = &W[(size_t)(n0 + tid) * 768 + k0];
            __bf16* dst = &Bs[sb][tid][0];
            async_cp16(src,      dst);
            async_cp16(src +  8, dst +  8);
            async_cp16(src + 16, dst + 16);
            async_cp16(src + 24, dst + 24);
        }
    };

    auto compute = [&](int sb) {
        v16bf af[2];
#pragma unroll
        for (int mi = 0; mi < 2; ++mi) {
            const int ar = 32 * wm + 16 * mi + l16;
            af[mi] = ld_frag(&As[sb][ar][8 * half], &As[sb][ar][16 + 8 * half]);
        }
#pragma unroll
        for (int ns = 0; ns < 4; ++ns) {
            const int br = 64 * wn + 16 * ns + l16;
            v16bf bfrag = ld_frag(&Bs[sb][br][8 * half], &Bs[sb][br][16 + 8 * half]);
#pragma unroll
            for (int mi = 0; mi < 2; ++mi)
                acc[mi][ns] = wmma_bf16(af[mi], bfrag, acc[mi][ns]);
        }
    };

    stage(0, 0);
    for (int it = 0; it < 24; ++it) {
        wait_async();
        __syncthreads();
        if (it + 1 < 24) stage((it + 1) & 1, (it + 1) * 32);
        compute(it & 1);
    }

#pragma unroll
    for (int ns = 0; ns < 4; ++ns) {
        const int n = n0 + 64 * wn + 16 * ns + l16;
        const int which = n / 768, rem = n % 768;
        const int nh = rem >> 6, d = rem & 63;
        __bf16* dst = (which == 0) ? qb : ((which == 1) ? kb : vb);
        const float bv = bias[n];
#pragma unroll
        for (int mi = 0; mi < 2; ++mi) {
#pragma unroll
            for (int j = 0; j < 8; ++j) {
                const int m = m0 + 32 * wm + 16 * mi + j + 8 * half;
                const int bidx = m / 196, hw = m % 196;
                dst[((((size_t)bidx * NHD + nh) * HWD + hw) * HDIM) + d] =
                    (__bf16)(acc[mi][ns][j] + bv);
            }
        }
    }
}

// ---------------------------------------------------------------------------
// Decomposed relative position bias: relh[bh][q][kh], relw[bh][q][kw] (f32)
// ---------------------------------------------------------------------------
__global__ __launch_bounds__(256)
void relpos_kernel(const __bf16* __restrict__ qbuf,
                   const float* __restrict__ rph, const float* __restrict__ rpw,
                   float* __restrict__ relh, float* __restrict__ relw) {
    const long long TOT = (long long)BH * HWD * 14;
    long long i = (long long)blockIdx.x * 256 + threadIdx.x;
    if (i >= 2 * TOT) return;
    const int sel = (i >= TOT);
    long long r = sel ? (i - TOT) : i;
    const int kk = (int)(r % 14);
    long long t = r / 14;
    const int q = (int)(t % HWD);
    const long long bh = t / HWD;
    const int qh = q / 14, qw = q % 14;
    const __bf16* qrow = qbuf + (bh * HWD + q) * HDIM;
    const float* tab = sel ? (rpw + (size_t)(qw - kk + 13) * HDIM)
                           : (rph + (size_t)(qh - kk + 13) * HDIM);
    float s = 0.f;
#pragma unroll 8
    for (int d = 0; d < HDIM; ++d) s += (float)qrow[d] * tab[d];
    (sel ? relw : relh)[r] = s;
}

// ---------------------------------------------------------------------------
// Fused attention: one block per (b, head). 256 threads = 8 waves.
// K (208x64, async-staged), V^T (64x224), per-(b,h) rel-pos tables and
// per-wave P staging all live in LDS (134.6 KB -> 2 workgroups/WGP).
// Each wave owns q-tiles {wave, wave+8}; full softmax row kept in 13 f32
// WMMA accumulators. Output bf16 to (b, hw, nh*64+d) = (78400 x 768).
// ---------------------------------------------------------------------------
__global__ __launch_bounds__(256)
void attn_kernel(const __bf16* __restrict__ qb, const __bf16* __restrict__ kb,
                 const __bf16* __restrict__ vb,
                 const float* __restrict__ relh, const float* __restrict__ relw,
                 __bf16* __restrict__ ob) {
    __shared__ __bf16 Kt[208][64];        // k-major
    __shared__ __bf16 Vt[64][224];        // d-major (transposed), zero-padded
    __shared__ __bf16 Ps[8][16][224];     // per-wave probability staging
    __shared__ float  Rh[HWD * 14];       // rel-pos bias tables for this (b,h)
    __shared__ float  Rw[HWD * 14];

    const int tid  = threadIdx.x;
    const int wave = tid >> 5, lane = tid & 31;
    const int half = lane >> 4, l16 = lane & 15;
    const int bh   = blockIdx.x;                 // 0..4799
    const size_t hbase = (size_t)bh * HWD * HDIM;
    const size_t rbase = (size_t)bh * HWD * 14;
    const int bidx = bh / NHD, nh = bh % NHD;
    const float scale = 0.125f;                  // 64^-0.5

    // async-stage K rows 0..195 (196*64 bf16 = 1568 x 16B chunks)
    for (int ch = tid; ch < 1568; ch += 256) {
        const int k = ch >> 3, c = (ch & 7) * 8;
        async_cp16(&kb[hbase + (size_t)k * HDIM + c], &Kt[k][c]);
    }
    // zero-pad K rows 196..207 (12 rows x 16 chunks of 4 bf16)
    if (tid < 192) {
        const int k = 196 + (tid >> 4), c = (tid & 15) * 4;
        *(uint2*)&Kt[k][c] = make_uint2(0u, 0u);
    }
    // stage V transposed
    for (int it = tid; it < 208 * 16; it += 256) {
        const int k = it >> 4, c = (it & 15) * 4;
        if (k < HWD) {
            uint2 val = *(const uint2*)&vb[hbase + (size_t)k * HDIM + c];
            const __bf16* pv = (const __bf16*)&val;
            Vt[c + 0][k] = pv[0]; Vt[c + 1][k] = pv[1];
            Vt[c + 2][k] = pv[2]; Vt[c + 3][k] = pv[3];
        } else {
            const __bf16 z = (__bf16)0.f;
            Vt[c + 0][k] = z; Vt[c + 1][k] = z; Vt[c + 2][k] = z; Vt[c + 3][k] = z;
        }
    }
    for (int it = tid; it < 64 * 16; it += 256) {          // cols 208..223
        Vt[it >> 4][208 + (it & 15)] = (__bf16)0.f;
    }
    // stage rel-pos bias tables
    for (int i = tid; i < HWD * 14; i += 256) {
        Rh[i] = relh[rbase + i];
        Rw[i] = relw[rbase + i];
    }
    wait_async();
    __syncthreads();

    for (int pass = 0; pass < 2; ++pass) {
        const int qt = wave + 8 * pass;
        if (qt >= 13) continue;
        const int q0 = qt * 16;

        // Q fragments (16 x 64 = two 16x32 A-fragments), loaded from global
        const int qrow = q0 + l16;
        const int qcl  = (qrow < HWD) ? qrow : (HWD - 1);
        const __bf16* qp = qb + hbase + (size_t)qcl * HDIM;
        v16bf qa0 = ld_frag(qp + 8 * half,      qp + 16 + 8 * half);
        v16bf qa1 = ld_frag(qp + 32 + 8 * half, qp + 48 + 8 * half);

        // S = Q K^T   (13 column tiles x two K-chunks)
        v8f S[13];
#pragma unroll
        for (int t = 0; t < 13; ++t) {
            v8f s = {};
            const int kc = 16 * t + l16;
            v16bf b0 = ld_frag(&Kt[kc][8 * half],      &Kt[kc][16 + 8 * half]);
            s = wmma_bf16(qa0, b0, s);
            v16bf b1 = ld_frag(&Kt[kc][32 + 8 * half], &Kt[kc][48 + 8 * half]);
            s = wmma_bf16(qa1, b1, s);
            S[t] = s;
        }

        // scale + decomposed rel-pos bias + padding mask, track row max
        float rmax[8];
#pragma unroll
        for (int j = 0; j < 8; ++j) rmax[j] = -3.0e38f;
#pragma unroll
        for (int t = 0; t < 13; ++t) {
            const int k = 16 * t + l16;
            const int kh = k / 14, kw = k % 14;
#pragma unroll
            for (int j = 0; j < 8; ++j) {
                const int q = q0 + j + 8 * half;
                const int qc = (q < HWD) ? q : (HWD - 1);
                float s;
                if (k < HWD) {
                    s = S[t][j] * scale + Rh[qc * 14 + kh] + Rw[qc * 14 + kw];
                } else {
                    s = -1.0e30f;
                }
                S[t][j] = s;
                rmax[j] = fmaxf(rmax[j], s);
            }
        }
        // half-wave (16-lane) row reductions
        float rs[8];
#pragma unroll
        for (int j = 0; j < 8; ++j) {
            float m = rmax[j];
            m = fmaxf(m, __shfl_xor(m, 1));
            m = fmaxf(m, __shfl_xor(m, 2));
            m = fmaxf(m, __shfl_xor(m, 4));
            m = fmaxf(m, __shfl_xor(m, 8));
            rmax[j] = m;
            rs[j] = 0.f;
        }
        // exponentiate, accumulate row sums, stage P (bf16) in per-wave LDS
#pragma unroll
        for (int t = 0; t < 13; ++t) {
#pragma unroll
            for (int j = 0; j < 8; ++j) {
                const float p = __expf(S[t][j] - rmax[j]);
                rs[j] += p;
                Ps[wave][j + 8 * half][16 * t + l16] = (__bf16)p;
            }
        }
#pragma unroll
        for (int j = 0; j < 8; ++j)                       // zero pad cols 208..223
            Ps[wave][j + 8 * half][208 + l16] = (__bf16)0.f;
#pragma unroll
        for (int j = 0; j < 8; ++j) {
            float s = rs[j];
            s += __shfl_xor(s, 1);
            s += __shfl_xor(s, 2);
            s += __shfl_xor(s, 4);
            s += __shfl_xor(s, 8);
            rs[j] = s;
        }
        // same-wave LDS RAW: make P stores visible before fragment reloads
        asm volatile("s_wait_dscnt 0x0" ::: "memory");

        // O = P V
        v8f O[4] = {};
#pragma unroll
        for (int kc = 0; kc < 7; ++kc) {
            const __bf16* pr = &Ps[wave][l16][32 * kc];
            v16bf pa = ld_frag(pr + 8 * half, pr + 16 + 8 * half);
#pragma unroll
            for (int n = 0; n < 4; ++n) {
                const __bf16* vr = &Vt[16 * n + l16][32 * kc];
                v16bf vf = ld_frag(vr + 8 * half, vr + 16 + 8 * half);
                O[n] = wmma_bf16(pa, vf, O[n]);
            }
        }

        // normalize + store bf16 to (b, hw, nh*64+d)
#pragma unroll
        for (int n = 0; n < 4; ++n) {
            const int d = nh * HDIM + 16 * n + l16;
#pragma unroll
            for (int j = 0; j < 8; ++j) {
                const int q = q0 + j + 8 * half;
                if (q < HWD) {
                    const float o = O[n][j] * (1.0f / rs[j]);
                    ob[((size_t)bidx * HWD + q) * CC + d] = (__bf16)o;
                }
            }
        }
    }
}

// ---------------------------------------------------------------------------
// Output projection: A(78400x768,bf16) @ Wp(768x768,bf16 N-major)^T + bias -> f32
// Same structure as qkv_gemm (64x256 block tile, double-buffered async LDS).
// ---------------------------------------------------------------------------
__global__ __launch_bounds__(256)
void proj_gemm(const __bf16* __restrict__ A, const __bf16* __restrict__ W,
               const float* __restrict__ bias, float* __restrict__ out) {
    __shared__ __bf16 As[2][64][32];
    __shared__ __bf16 Bs[2][256][32];
    const int tid  = threadIdx.x;
    const int wave = tid >> 5, lane = tid & 31;
    const int half = lane >> 4, l16 = lane & 15;
    const int m0 = (blockIdx.x % 1225) * 64;
    const int n0 = (blockIdx.x / 1225) * 256;
    const int wm = wave >> 2, wn = wave & 3;

    v8f acc[2][4] = {};

    auto stage = [&](int sb, int k0) {
        {
            const int r = tid >> 2, c = (tid & 3) * 8;
            async_cp16(&A[(size_t)(m0 + r) * 768 + k0 + c], &As[sb][r][c]);
        }
        {
            const __bf16* src = &W[(size_t)(n0 + tid) * 768 + k0];
            __bf16* dst = &Bs[sb][tid][0];
            async_cp16(src,      dst);
            async_cp16(src +  8, dst +  8);
            async_cp16(src + 16, dst + 16);
            async_cp16(src + 24, dst + 24);
        }
    };

    auto compute = [&](int sb) {
        v16bf af[2];
#pragma unroll
        for (int mi = 0; mi < 2; ++mi) {
            const int ar = 32 * wm + 16 * mi + l16;
            af[mi] = ld_frag(&As[sb][ar][8 * half], &As[sb][ar][16 + 8 * half]);
        }
#pragma unroll
        for (int ns = 0; ns < 4; ++ns) {
            const int br = 64 * wn + 16 * ns + l16;
            v16bf bfrag = ld_frag(&Bs[sb][br][8 * half], &Bs[sb][br][16 + 8 * half]);
#pragma unroll
            for (int mi = 0; mi < 2; ++mi)
                acc[mi][ns] = wmma_bf16(af[mi], bfrag, acc[mi][ns]);
        }
    };

    stage(0, 0);
    for (int it = 0; it < 24; ++it) {
        wait_async();
        __syncthreads();
        if (it + 1 < 24) stage((it + 1) & 1, (it + 1) * 32);
        compute(it & 1);
    }

#pragma unroll
    for (int ns = 0; ns < 4; ++ns) {
        const int n = n0 + 64 * wn + 16 * ns + l16;
        const float bv = bias[n];
#pragma unroll
        for (int mi = 0; mi < 2; ++mi) {
#pragma unroll
            for (int j = 0; j < 8; ++j) {
                const int m = m0 + 32 * wm + 16 * mi + j + 8 * half;
                out[(size_t)m * 768 + n] = acc[mi][ns][j] + bv;
            }
        }
    }
}

// ---------------------------------------------------------------------------
// Host launcher
// ---------------------------------------------------------------------------
extern "C" void kernel_launch(void* const* d_in, const int* in_sizes, int n_in,
                              void* d_out, int out_size, void* d_ws, size_t ws_size,
                              hipStream_t stream) {
    (void)in_sizes; (void)n_in; (void)out_size; (void)ws_size;
    const float* hidden = (const float*)d_in[0];
    const float* qkv_w  = (const float*)d_in[1];
    const float* qkv_b  = (const float*)d_in[2];
    const float* proj_w = (const float*)d_in[3];
    const float* proj_b = (const float*)d_in[4];
    const float* rph    = (const float*)d_in[5];
    const float* rpw    = (const float*)d_in[6];
    float* out = (float*)d_out;

    char* ws = (char*)d_ws;
    size_t off = 0;
    auto carve = [&](size_t bytes) {
        char* p = ws + off;
        off += (bytes + 255) & ~(size_t)255;
        return p;
    };
    __bf16* Xbf  = (__bf16*)carve((size_t)MTOT * CC * 2);
    __bf16* Wq   = (__bf16*)carve((size_t)3 * CC * CC * 2);
    __bf16* Wp   = (__bf16*)carve((size_t)CC * CC * 2);
    __bf16* qbuf = (__bf16*)carve((size_t)BH * HWD * HDIM * 2);
    __bf16* kbuf = (__bf16*)carve((size_t)BH * HWD * HDIM * 2);
    __bf16* vbuf = (__bf16*)carve((size_t)BH * HWD * HDIM * 2);
    float*  relh = (float*)carve((size_t)BH * HWD * 14 * 4);
    float*  relw = (float*)carve((size_t)BH * HWD * 14 * 4);
    __bf16* abuf = (__bf16*)carve((size_t)MTOT * CC * 2);

    // f32 -> bf16 conversions
    {
        long long n4 = (long long)MTOT * CC / 4;
        cvt_f32_bf16<<<(unsigned)((n4 + 255) / 256), 256, 0, stream>>>(hidden, Xbf, n4);
    }
    {
        long long n4 = (long long)3 * CC * CC / 4;
        cvt_f32_bf16<<<(unsigned)((n4 + 255) / 256), 256, 0, stream>>>(qkv_w, Wq, n4);
    }
    {
        long long n4 = (long long)CC * CC / 4;
        cvt_f32_bf16<<<(unsigned)((n4 + 255) / 256), 256, 0, stream>>>(proj_w, Wp, n4);
    }

    // QKV projection (M=78400, N=2304): 1225 x 9 tiles
    qkv_gemm<<<1225 * 9, 256, 0, stream>>>(Xbf, Wq, qkv_b, qbuf, kbuf, vbuf);

    // rel-pos bias tables
    {
        long long tot2 = 2LL * BH * HWD * 14;
        relpos_kernel<<<(unsigned)((tot2 + 255) / 256), 256, 0, stream>>>(
            qbuf, rph, rpw, relh, relw);
    }

    // fused attention: one block per (b, head)
    attn_kernel<<<BH, 256, 0, stream>>>(qbuf, kbuf, vbuf, relh, relw, abuf);

    // output projection (M=78400, N=768): 1225 x 3 tiles
    proj_gemm<<<1225 * 3, 256, 0, stream>>>(abuf, Wp, proj_b, out);
}
